// GeneratorRNN_64467459113442
// MI455X (gfx1250) — matrix operand order repaired and checked by reading, not compiled, not run on hardware
//
#include <hip/hip_runtime.h>

// ---------------------------------------------------------------------------
// LSTM forward on MI455X (gfx1250): bf16 WMMA, f32 accumulate.
// Phase 0: pack/convert inputs to bf16, fuse biases.
// Phase 1: x_gates[T*B,4096] = Xbf[T*B,768] @ Wih^T  (WMMA tiled GEMM, bf16 out)
// Phase 2: persistent kernel, 512 sequential steps; W_hh tile staged ONCE into
//          LDS via the Tensor Data Mover, c-state in registers, h double-
//          buffered in L2-resident global, device-wide barrier per step.
// Phase 3: out[128,256] = h @ fc_w^T + fc_b (WMMA).
// ---------------------------------------------------------------------------

typedef __attribute__((ext_vector_type(16))) __bf16 v16bf;
typedef __attribute__((ext_vector_type(8)))  float  v8f;
typedef unsigned __attribute__((ext_vector_type(4))) u32x4;
typedef int      __attribute__((ext_vector_type(4))) i32x4;
typedef int      __attribute__((ext_vector_type(8))) i32x8;

#define OBS_DIM 512
#define ACTD    256
#define IN_DIM  768
#define HID     1024
#define G4      4096
#define BATCH   128
#define TLEN    512
#define LSTM_BLOCKS 64
// LDS B layout: 64 rows (4 gates x 16 cols), each 512 dwords + 2 pad dwords
// per 256 dwords => 516 dwords/row. Breaks the power-of-two bank stride.
#define BROW_DW 516
#define LDS_B_BYTES (64 * BROW_DW * 4)

#if defined(__has_builtin)
#if __has_builtin(__builtin_amdgcn_tensor_load_to_lds) && \
    __has_builtin(__builtin_amdgcn_s_wait_tensorcnt)
#define HAVE_TDM 1
#endif
#endif

union FragU { v16bf v; unsigned u32[8]; };

__device__ __forceinline__ float bf2f(unsigned short h) {
    return __uint_as_float(((unsigned)h) << 16);
}
__device__ __forceinline__ unsigned short f2bf(float f) {
    unsigned u = __float_as_uint(f);
    u += 0x7FFFu + ((u >> 16) & 1u);           // round to nearest even
    return (unsigned short)(u >> 16);
}
__device__ __forceinline__ float sigm(float x) { return 1.0f / (1.0f + __expf(-x)); }
__device__ __forceinline__ float tanh_fast(float x) {
    float e = __expf(2.0f * x);
    return 1.0f - 2.0f / (e + 1.0f);
}

// ---- Phase 0 kernels -------------------------------------------------------

__global__ void k_pack_x(const float* __restrict__ obs, const float* __restrict__ act,
                         unsigned short* __restrict__ xb) {
    long long i = (long long)blockIdx.x * blockDim.x + threadIdx.x;
    const long long total = (long long)TLEN * BATCH * IN_DIM;
    if (i >= total) return;
    int d = (int)(i % IN_DIM);
    long long r = i / IN_DIM;
    int b = (int)(r % BATCH);
    int t = (int)(r / BATCH);
    float v;
    if (d < OBS_DIM) v = obs[((long long)b * TLEN + t) * OBS_DIM + d];
    else             v = act[((long long)b * TLEN + t) * ACTD + (d - OBS_DIM)];
    xb[i] = f2bf(v);
}

__global__ void k_cvt_bf16(const float* __restrict__ src, unsigned short* __restrict__ dst,
                           long long n) {
    long long i = (long long)blockIdx.x * blockDim.x + threadIdx.x;
    if (i < n) dst[i] = f2bf(src[i]);
}

__global__ void k_bias_sum(const float* __restrict__ a, const float* __restrict__ b,
                           float* __restrict__ dst, int n) {
    int i = blockIdx.x * blockDim.x + threadIdx.x;
    if (i < n) dst[i] = a[i] + b[i];
}

// ---- Phase 1: x_gates GEMM -------------------------------------------------
__global__ void k_xgates(const unsigned short* __restrict__ Xb,
                         const unsigned short* __restrict__ Wb,
                         const float* __restrict__ bias,
                         unsigned short* __restrict__ xg) {
    const int lane = threadIdx.x & 31;
    const int wv   = threadIdx.x >> 5;
    const long long gid = (long long)blockIdx.x * 8 + wv;    // 4096*256 tiles
    const int mt = (int)(gid >> 8);
    const int n0 = (int)(gid & 255) * 16;
    const int ncol = lane & 15;
    const int hi   = lane >> 4;

    v8f acc;
    const float bv = bias[n0 + ncol];
#pragma unroll
    for (int v = 0; v < 8; ++v) acc[v] = bv;

    const int aKb = hi ? 16 : 0;
    const int bKb = hi ? 16 : 0;
    const unsigned* Arow = (const unsigned*)(Xb + (long long)(mt * 16 + ncol) * IN_DIM);
    const unsigned* Brow = (const unsigned*)(Wb + (long long)(n0 + ncol) * IN_DIM);

    for (int k0 = 0; k0 < IN_DIM; k0 += 32) {
        FragU a, b;
#pragma unroll
        for (int v = 0; v < 4; ++v) {
            a.u32[v]     = Arow[(k0 + aKb + 2 * v) >> 1];
            a.u32[v + 4] = Arow[(k0 + 8 + aKb + 2 * v) >> 1];
        }
#pragma unroll
        for (int v = 0; v < 8; ++v) b.u32[v] = Brow[(k0 + bKb + 2 * v) >> 1];
        acc = __builtin_amdgcn_wmma_f32_16x16x32_bf16(false, a.v, false, b.v,
                                                      (short)0, acc, false, false);
    }
    unsigned short* out = xg + (long long)(mt * 16 + hi * 8) * G4 + n0 + ncol;
#pragma unroll
    for (int v = 0; v < 8; ++v) out[(long long)v * G4] = f2bf(acc[v]);
}

// ---- Phase 2: persistent recurrent kernel ----------------------------------
// nt = blockIdx (hidden tile), mt = wave-in-block (batch tile): the block's
// 8 waves share one 128KB W_hh slice, staged once into LDS via the TDM.
__global__ void __launch_bounds__(256, 1)
k_lstm(const unsigned short* __restrict__ xg,
       const unsigned short* __restrict__ Whh,
       unsigned short* __restrict__ h0,
       unsigned short* __restrict__ h1,
       unsigned* __restrict__ barrier) {
    extern __shared__ unsigned lds_b[];

    const int lane = threadIdx.x & 31;
    const int wv   = threadIdx.x >> 5;
    const int mt = wv;               // 0..7   batch tile
    const int nt = blockIdx.x;       // 0..63  hidden tile
    const int m0 = mt * 16, n0 = nt * 16;
    const int ncol = lane & 15;
    const int hi   = lane >> 4;
    const int aKb = hi ? 16 : 0;
    const int bKb = hi ? 16 : 0;

    // ---- one-time W_hh staging: global -> LDS (TDM if available) ----------
#ifdef HAVE_TDM
    if (wv < 4) {
        const int g = wv;
        unsigned long long ga =
            (unsigned long long)(uintptr_t)(Whh + ((size_t)(g * HID + n0)) * HID);
        unsigned ldsa = (unsigned)(uintptr_t)(&lds_b[(size_t)g * 16 * BROW_DW]);
        u32x4 g0;
        g0[0] = 1u;                                            // count=1
        g0[1] = ldsa;                                          // lds_addr
        g0[2] = (unsigned)(ga & 0xFFFFFFFFull);                // global_addr lo
        g0[3] = (unsigned)((ga >> 32) & 0x01FFFFFFull) | (2u << 30); // hi + type=2
        i32x8 g1;
        g1[0] = (int)((1u << 16) |   // data_size = 2 bytes
                      (1u << 20) |   // pad_enable
                      (7u << 22) |   // pad_interval: 256 dwords
                      (1u << 25));   // pad_amount: 2 dwords
        g1[1] = (int)(1024u << 16);  // tensor_dim0 = 1024 (bits 79:48, low part)
        g1[2] = (int)(16u << 16);    // tensor_dim1 = 16   (bits 111:80, low part)
        g1[3] = (int)(1024u << 16);  // tile_dim0 = 1024   (bits 127:112)
        g1[4] = 16;                  // tile_dim1 = 16; tile_dim2 = 0
        g1[5] = 1024;                // tensor_dim0_stride = 1024
        g1[6] = 0;
        g1[7] = 0;
        i32x4 z4 = {0, 0, 0, 0};
#if __clang_major__ >= 23
        i32x8 z8 = {0, 0, 0, 0, 0, 0, 0, 0};
        __builtin_amdgcn_tensor_load_to_lds(g0, g1, z4, z4, z8, 0);
#else
        __builtin_amdgcn_tensor_load_to_lds(g0, g1, z4, z4, 0);
#endif
        __builtin_amdgcn_s_wait_tensorcnt((short)0);
    }
#else
    // cooperative fallback: 64 rows x 512 dwords, padded layout
    for (int idx = threadIdx.x; idx < 64 * 512; idx += 256) {
        int R = idx >> 9, d = idx & 511;
        int wrow = (R >> 4) * HID + n0 + (R & 15);
        lds_b[R * BROW_DW + d + 2 * (d >> 8)] =
            ((const unsigned*)Whh)[(size_t)wrow * (HID / 2) + d];
    }
#endif
    __syncthreads();

    // precomputed LDS row bases for the 4 gates
    int Rb[4];
#pragma unroll
    for (int g = 0; g < 4; ++g) Rb[g] = (g * 16 + ncol) * BROW_DW;

    float c[8];
#pragma unroll
    for (int v = 0; v < 8; ++v) c[v] = 0.0f;

    unsigned short* hb[2] = { h0, h1 };

    for (int t = 0; t < TLEN; ++t) {
        const unsigned short* hcur = hb[t & 1];
        unsigned short* hnxt = hb[(t + 1) & 1];

        // init accumulators from precomputed input gates (bf16 -> f32)
        v8f acc[4];
        const unsigned short* xgrow = xg + (long long)t * BATCH * G4;
#pragma unroll
        for (int g = 0; g < 4; ++g) {
#pragma unroll
            for (int v = 0; v < 8; ++v) {
                int m = m0 + hi * 8 + v;
                acc[g][v] = bf2f(xgrow[(long long)m * G4 + g * HID + n0 + ncol]);
            }
        }

        // prefetch next timestep's x_gates rows (global_prefetch_b8)
        if (t + 1 < TLEN) {
            const unsigned short* xn =
                xg + (long long)(t + 1) * BATCH * G4 + (long long)(m0 + ncol) * G4;
#pragma unroll
            for (int g = 0; g < 4; ++g)
                __builtin_prefetch(xn + g * HID + n0, 0, 3);
        }

        const unsigned* Arow = (const unsigned*)(hcur + (long long)(m0 + ncol) * HID);
        for (int k0 = 0; k0 < HID; k0 += 32) {
            FragU a;
#pragma unroll
            for (int v = 0; v < 4; ++v) {
                a.u32[v]     = Arow[(k0 + aKb + 2 * v) >> 1];
                a.u32[v + 4] = Arow[(k0 + 8 + aKb + 2 * v) >> 1];
            }
            const int kd  = (k0 + bKb) >> 1;
            const int adj = kd + 2 * (kd >> 8);   // padded LDS dword offset
#pragma unroll
            for (int g = 0; g < 4; ++g) {
                FragU b;
#pragma unroll
                for (int v = 0; v < 8; ++v) b.u32[v] = lds_b[Rb[g] + adj + v];
                acc[g] = __builtin_amdgcn_wmma_f32_16x16x32_bf16(false, a.v, false, b.v,
                                                                 (short)0, acc[g], false, false);
            }
        }

        // fused gate nonlinearities + cell update; h stored bf16 for next step
        unsigned short* hout = hnxt + (long long)(m0 + hi * 8) * HID + n0 + ncol;
#pragma unroll
        for (int v = 0; v < 8; ++v) {
            float iv = sigm(acc[0][v]);
            float fv = sigm(acc[1][v]);
            float gv = tanh_fast(acc[2][v]);
            float ov = sigm(acc[3][v]);
            c[v] = fv * c[v] + iv * gv;
            hout[(long long)v * HID] = f2bf(ov * tanh_fast(c[v]));
        }

        // device-wide barrier: all h writes of step t visible before step t+1
        __threadfence();
        __syncthreads();
        if (threadIdx.x == 0) {
            __hip_atomic_fetch_add(barrier, 1u, __ATOMIC_RELEASE, __HIP_MEMORY_SCOPE_AGENT);
            const unsigned target = (unsigned)gridDim.x * (unsigned)(t + 1);
            while (__hip_atomic_load(barrier, __ATOMIC_ACQUIRE, __HIP_MEMORY_SCOPE_AGENT) < target)
                __builtin_amdgcn_s_sleep(1);
        }
        __syncthreads();
    }
}

// ---- Phase 3: final projection ---------------------------------------------
__global__ void k_fc(const unsigned short* __restrict__ hfin,
                     const unsigned short* __restrict__ fcw,
                     const float* __restrict__ fcb,
                     float* __restrict__ out) {
    const int lane = threadIdx.x & 31;
    const int wv   = threadIdx.x >> 5;
    const int waveId = blockIdx.x * 8 + wv;   // 8 x 16 = 128 tiles
    const int mt = waveId >> 4;
    const int n0 = (waveId & 15) * 16;
    const int ncol = lane & 15;
    const int hi   = lane >> 4;
    const int aKb = hi ? 16 : 0;
    const int bKb = hi ? 16 : 0;

    v8f acc;
    const float bv = fcb[n0 + ncol];
#pragma unroll
    for (int v = 0; v < 8; ++v) acc[v] = bv;

    const unsigned* Arow = (const unsigned*)(hfin + (long long)(mt * 16 + ncol) * HID);
    const unsigned* Brow = (const unsigned*)(fcw + (long long)(n0 + ncol) * HID);
    for (int k0 = 0; k0 < HID; k0 += 32) {
        FragU a, b;
#pragma unroll
        for (int v = 0; v < 4; ++v) {
            a.u32[v]     = Arow[(k0 + aKb + 2 * v) >> 1];
            a.u32[v + 4] = Arow[(k0 + 8 + aKb + 2 * v) >> 1];
        }
#pragma unroll
        for (int v = 0; v < 8; ++v) b.u32[v] = Brow[(k0 + bKb + 2 * v) >> 1];
        acc = __builtin_amdgcn_wmma_f32_16x16x32_bf16(false, a.v, false, b.v,
                                                      (short)0, acc, false, false);
    }
    float* o = out + (long long)(mt * 16 + hi * 8) * ACTD + n0 + ncol;
#pragma unroll
    for (int v = 0; v < 8; ++v) o[(long long)v * ACTD] = acc[v];
}

// ---- Host launcher ----------------------------------------------------------
extern "C" void kernel_launch(void* const* d_in, const int* in_sizes, int n_in,
                              void* d_out, int out_size, void* d_ws, size_t ws_size,
                              hipStream_t stream) {
    (void)in_sizes; (void)n_in; (void)out_size; (void)ws_size;
    const float* obs  = (const float*)d_in[0];
    const float* act  = (const float*)d_in[1];
    const float* Wih  = (const float*)d_in[2];
    const float* Whh  = (const float*)d_in[3];
    const float* bih  = (const float*)d_in[4];
    const float* bhh  = (const float*)d_in[5];
    const float* fcw  = (const float*)d_in[6];
    const float* fcb  = (const float*)d_in[7];
    float* out = (float*)d_out;

    char* ws = (char*)d_ws;
    size_t cur = 0;
    auto alloc = [&](size_t bytes) { char* p = ws + cur; cur = (cur + bytes + 255) & ~(size_t)255; return p; };
    unsigned short* xb    = (unsigned short*)alloc((size_t)TLEN * BATCH * IN_DIM * 2); // 96 MiB
    unsigned short* xg    = (unsigned short*)alloc((size_t)TLEN * BATCH * G4 * 2);     // 512 MiB
    unsigned short* wih_b = (unsigned short*)alloc((size_t)G4 * IN_DIM * 2);           // 6 MiB
    unsigned short* whh_b = (unsigned short*)alloc((size_t)G4 * HID * 2);              // 8 MiB
    unsigned short* fcw_b = (unsigned short*)alloc((size_t)ACTD * HID * 2);            // 0.5 MiB
    float*          bias  = (float*)alloc((size_t)G4 * 4);
    unsigned short* h0    = (unsigned short*)alloc((size_t)BATCH * HID * 2);
    unsigned short* h1    = (unsigned short*)alloc((size_t)BATCH * HID * 2);
    unsigned*       bar   = (unsigned*)alloc(256);

    hipMemsetAsync(h0, 0, (size_t)BATCH * HID * 2, stream);
    hipMemsetAsync(bar, 0, 256, stream);

    // allow 132KB of dynamic LDS for the persistent kernel
    hipFuncSetAttribute((const void*)k_lstm,
                        hipFuncAttributeMaxDynamicSharedMemorySize, LDS_B_BYTES);

    const int TPB = 256;
    {
        long long n = (long long)TLEN * BATCH * IN_DIM;
        k_pack_x<<<(unsigned)((n + TPB - 1) / TPB), TPB, 0, stream>>>(obs, act, xb);
    }
    {
        long long n = (long long)G4 * IN_DIM;
        k_cvt_bf16<<<(unsigned)((n + TPB - 1) / TPB), TPB, 0, stream>>>(Wih, wih_b, n);
    }
    {
        long long n = (long long)G4 * HID;
        k_cvt_bf16<<<(unsigned)((n + TPB - 1) / TPB), TPB, 0, stream>>>(Whh, whh_b, n);
    }
    {
        long long n = (long long)ACTD * HID;
        k_cvt_bf16<<<(unsigned)((n + TPB - 1) / TPB), TPB, 0, stream>>>(fcw, fcw_b, n);
    }
    k_bias_sum<<<(G4 + TPB - 1) / TPB, TPB, 0, stream>>>(bih, bhh, bias, G4);

    k_xgates<<<(4096 * 256) / 8, TPB, 0, stream>>>(xb, wih_b, bias, xg);

    k_lstm<<<LSTM_BLOCKS, TPB, LDS_B_BYTES, stream>>>(xg, whh_b, h0, h1, bar);

    k_fc<<<128 / 8, TPB, 0, stream>>>(h0, fcw_b, fcb, out);
}
